// MeanEmbeddingMatcher_68831145886145
// MI455X (gfx1250) — compile-verified, matrix operand order checked by reading, not run on hardware
//
#include <hip/hip_runtime.h>
#include <hip/hip_bf16.h>
#include <math.h>

// ---- CDNA5 WMMA vector types ----
typedef __attribute__((ext_vector_type(16))) __bf16 v16bf;
typedef __attribute__((ext_vector_type(8)))  __bf16 v8bf;
typedef __attribute__((ext_vector_type(8)))  float  v8f;

#define B_ROWS 2048
#define C_ROWS 100000
#define DIM    512
#define TOPK   100
#define NSEG   16
#define SEGC   ((C_ROWS + NSEG - 1) / NSEG)   // 6250 columns per segment
#define MBLOCK 64
#define NWAVES 8
#define MTILES (MBLOCK / 16)                  // 4 M-tiles per wave
#define NT     2                              // N-tiles per wave (B-fragment reuse)
#define NSTRIP (NWAVES * NT)                  // 16 column strips of 16
#define NITER_COLS (NSTRIP * 16)              // 256 columns per outer iteration
#define ASTRIDE (DIM + 8)                     // padded LDS row stride (bf16 elems)
#define NCAND  (NSEG * TOPK)                  // 1600 merge candidates per row
#define SORTN  2048                           // bitonic pad

#define NEG_INF (-__builtin_inff())

// ---------------------------------------------------------------------------
// Kernel 1: L2-normalize rows (fp32 -> bf16). NaN rows stay NaN:
//   fmaxf(NaN, eps) = eps, but x is NaN so x*inv stays NaN.
// ---------------------------------------------------------------------------
__global__ void __launch_bounds__(256)
nrm_bf16_kernel(const float* __restrict__ src, __bf16* __restrict__ dst) {
  const int row = blockIdx.x;
  const float* x = src + (size_t)row * DIM;
  __shared__ float red[256];
  float s = 0.0f;
  for (int k = threadIdx.x; k < DIM; k += 256) { float v = x[k]; s += v * v; }
  red[threadIdx.x] = s;
  __syncthreads();
  for (int off = 128; off > 0; off >>= 1) {
    if (threadIdx.x < off) red[threadIdx.x] += red[threadIdx.x + off];
    __syncthreads();
  }
  const float inv = 1.0f / fmaxf(sqrtf(red[0]), 1e-12f);
  __bf16* y = dst + (size_t)row * DIM;
  for (int k = threadIdx.x; k < DIM; k += 256) y[k] = (__bf16)(x[k] * inv);
}

// ---------------------------------------------------------------------------
// Kernel 2: fused bf16-WMMA GEMM + streaming per-row top-K.
// Block = 256 threads (8 waves), 64 rows x one column segment per block.
// Per K-step each wave runs a dense group of NT*MTILES = 8
// v_wmma_f32_16x16x32_bf16. B fragments use ping-pong register
// double-buffering (K unrolled by 2) so no register rotation -> no
// WMMA->VALU hazard NOPs in the hot loop.
// ---------------------------------------------------------------------------
__global__ void __launch_bounds__(256, 1)
gemm_topk_kernel(const __bf16* __restrict__ Abf, const __bf16* __restrict__ Gbf,
                 float* __restrict__ cand_val, int* __restrict__ cand_idx) {
  extern __shared__ char smem[];
  __bf16* Atile  = (__bf16*)smem;                                        // 64*520 bf16
  float*  staged = (float*)(smem + MBLOCK * ASTRIDE * sizeof(__bf16));   // 16*4*256 f32
  float*  topkv  = staged + NSTRIP * MTILES * 256;                       // 64*100 f32
  int*    topki  = (int*)(topkv + MBLOCK * TOPK);                        // 64*100 i32
  float*  rowmin = (float*)(topki + MBLOCK * TOPK);                      // 64 f32
  int*    rowpos = (int*)(rowmin + MBLOCK);                              // 64 i32

  const int tid  = threadIdx.x;
  const int wave = tid >> 5;
  const int lane = tid & 31;
  const int m0   = blockIdx.x * MBLOCK;
  const int seg  = blockIdx.y;
  const int segStart = seg * SEGC;
  const int segEnd   = min(segStart + SEGC, C_ROWS);

  // Stage normalized A rows into LDS (padded stride kills bank conflicts).
  for (int idx = tid * 8; idx < MBLOCK * DIM; idx += 256 * 8) {
    const int m = idx >> 9, k = idx & (DIM - 1);
    *(v8bf*)&Atile[m * ASTRIDE + k] = *(const v8bf*)&Abf[(size_t)(m0 + m) * DIM + k];
  }
  for (int i = tid; i < MBLOCK * TOPK; i += 256) { topkv[i] = NEG_INF; topki[i] = -1; }
  for (int i = tid; i < MBLOCK; i += 256)        { rowmin[i] = NEG_INF; rowpos[i] = 0; }
  __syncthreads();

  const int laneM = lane & 15;   // A row / B column within tile
  const int laneG = lane >> 4;   // K half-group per ISA 16-bit fragment layout
  const int kOff  = 8 * laneG;   // lane's K sub-offset within a 32-wide K step

  for (int n0 = segStart; n0 < segEnd; n0 += NITER_COLS) {
    // Per-lane B row base pointers for this wave's NT column tiles.
    const __bf16* bbase[NT];
#pragma unroll
    for (int nt = 0; nt < NT; ++nt) {
      const int cw = n0 + (wave * NT + nt) * 16 + laneM;
      bbase[nt] = Gbf + (size_t)min(cw, C_ROWS - 1) * DIM;  // clamp tail, filtered later
    }

    v8f acc[NT][MTILES] = {};

    // Dense 8-WMMA group at K offset ktg using the given B register buffer.
    auto wmma_group = [&](int ktg, const v8bf* bl, const v8bf* bh) {
      v16bf af[MTILES];
#pragma unroll
      for (int mt = 0; mt < MTILES; ++mt) {
        const __bf16* ap = &Atile[(mt * 16 + laneM) * ASTRIDE + ktg + kOff];
        const v8bf alo = *(const v8bf*)ap;
        const v8bf ahi = *(const v8bf*)(ap + 16);
        af[mt] = __builtin_shufflevector(alo, ahi,
            0,1,2,3,4,5,6,7,8,9,10,11,12,13,14,15);
      }
      v16bf bf[NT];
#pragma unroll
      for (int nt = 0; nt < NT; ++nt)
        bf[nt] = __builtin_shufflevector(bl[nt], bh[nt],
            0,1,2,3,4,5,6,7,8,9,10,11,12,13,14,15);
#pragma unroll
      for (int nt = 0; nt < NT; ++nt)
#pragma unroll
        for (int mt = 0; mt < MTILES; ++mt)
          acc[nt][mt] = __builtin_amdgcn_wmma_f32_16x16x32_bf16(
              false, af[mt], false, bf[nt], (short)0, acc[nt][mt], false, false);
    };

    // Ping-pong B double-buffer, K unrolled by 2 (steps of 64).
    v8bf blo0[NT], bhi0[NT], blo1[NT], bhi1[NT];
#pragma unroll
    for (int nt = 0; nt < NT; ++nt) {
      blo0[nt] = *(const v8bf*)(bbase[nt] + kOff);
      bhi0[nt] = *(const v8bf*)(bbase[nt] + kOff + 16);
    }
    for (int kt = 0; kt < DIM; kt += 64) {
      // Prefetch buffer 1 (kt+32; always in range since DIM % 64 == 0).
#pragma unroll
      for (int nt = 0; nt < NT; ++nt) {
        blo1[nt] = *(const v8bf*)(bbase[nt] + kt + 32 + kOff);
        bhi1[nt] = *(const v8bf*)(bbase[nt] + kt + 32 + kOff + 16);
      }
      wmma_group(kt, blo0, bhi0);
      // Prefetch buffer 0 for the next double-step.
      if (kt + 64 < DIM) {
#pragma unroll
        for (int nt = 0; nt < NT; ++nt) {
          blo0[nt] = *(const v8bf*)(bbase[nt] + kt + 64 + kOff);
          bhi0[nt] = *(const v8bf*)(bbase[nt] + kt + 64 + kOff + 16);
        }
      }
      wmma_group(kt + 32, blo1, bhi1);
    }

    // Epilogue: D layout -> LDS staging, NaN -> -inf (reference semantics).
#pragma unroll
    for (int nt = 0; nt < NT; ++nt)
#pragma unroll
      for (int mt = 0; mt < MTILES; ++mt)
#pragma unroll
        for (int r = 0; r < 8; ++r) {
          float v = acc[nt][mt][r];
          v = (v != v) ? NEG_INF : v;
          const int mloc = r + 8 * laneG;
          staged[((wave * NT + nt) * MTILES + mt) * 256 + mloc * 16 + laneM] = v;
        }
    __syncthreads();

    // Streaming top-K: one thread per row, threshold-gated replace-min.
    if (tid < MBLOCK) {
      const int t = tid;
      float rmin = rowmin[t];
      int   rpos = rowpos[t];
      float* tv = topkv + t * TOPK;
      int*   ti = topki + t * TOPK;
      const int mtile = t >> 4, mloc = t & 15;
      for (int s = 0; s < NSTRIP; ++s) {
        const float* src = staged + (s * MTILES + mtile) * 256 + mloc * 16;
        const int cbase = n0 + s * 16;
        for (int n = 0; n < 16; ++n) {
          const int c = cbase + n;
          const float v = src[n];
          if (c < segEnd && v > rmin) {
            tv[rpos] = v; ti[rpos] = c;
            rmin = tv[0]; rpos = 0;
            for (int j = 1; j < TOPK; ++j) {
              const float q = tv[j];
              if (q < rmin) { rmin = q; rpos = j; }
            }
          }
        }
      }
      rowmin[t] = rmin; rowpos[t] = rpos;
    }
    __syncthreads();
  }

  // Emit this segment's (unsorted) per-row candidates.
  for (int i = tid; i < MBLOCK * TOPK; i += 256) {
    const int t = i / TOPK, j = i - t * TOPK;
    const size_t o = ((size_t)(m0 + t) * NSEG + seg) * TOPK + j;
    cand_val[o] = topkv[t * TOPK + j];
    cand_idx[o] = topki[t * TOPK + j];
  }
}

// ---------------------------------------------------------------------------
// Kernel 3: per-row merge of 16x100 candidates -> sorted top-100.
// Bitonic sort of 2048 (pad -inf), order: value desc, index asc (jax tie rule).
// ---------------------------------------------------------------------------
__global__ void __launch_bounds__(256)
merge_topk_kernel(const float* __restrict__ cand_val, const int* __restrict__ cand_idx,
                  float* __restrict__ out_val, int* __restrict__ out_idx) {
  __shared__ float sv[SORTN];
  __shared__ int   si[SORTN];
  const int row = blockIdx.x, tid = threadIdx.x;
  for (int i = tid; i < SORTN; i += 256) {
    if (i < NCAND) {
      sv[i] = cand_val[(size_t)row * NCAND + i];
      si[i] = cand_idx[(size_t)row * NCAND + i];
    } else {
      sv[i] = NEG_INF; si[i] = 0x7fffffff;
    }
  }
  __syncthreads();
  for (int k = 2; k <= SORTN; k <<= 1) {
    for (int j = k >> 1; j > 0; j >>= 1) {
      for (int i = tid; i < SORTN; i += 256) {
        const int ixj = i ^ j;
        if (ixj > i) {
          const float va = sv[i], vb = sv[ixj];
          const int   ia = si[i], ib = si[ixj];
          const bool aBetter = (va > vb) || (va == vb && ia < ib);
          const bool bBetter = (vb > va) || (vb == va && ib < ia);
          const bool desc = ((i & k) == 0);
          if (desc ? bBetter : aBetter) {
            sv[i] = vb; sv[ixj] = va; si[i] = ib; si[ixj] = ia;
          }
        }
      }
      __syncthreads();
    }
  }
  for (int i = tid; i < TOPK; i += 256) {
    out_val[(size_t)row * TOPK + i] = sv[i];
    out_idx[(size_t)row * TOPK + i] = si[i];
  }
}

// ---------------------------------------------------------------------------
extern "C" void kernel_launch(void* const* d_in, const int* in_sizes, int n_in,
                              void* d_out, int out_size, void* d_ws, size_t ws_size,
                              hipStream_t stream) {
  const float* emb = (const float*)d_in[0];   // [2048, 512] f32
  const float* idx = (const float*)d_in[1];   // [100000, 512] f32
  (void)in_sizes; (void)n_in; (void)out_size; (void)ws_size; // k==100 baked in

  // Workspace layout (~131 MB): Abf | Gbf | cand_val | cand_idx
  char* ws = (char*)d_ws;
  __bf16* Abf = (__bf16*)ws;
  size_t off = (size_t)B_ROWS * DIM * sizeof(__bf16);
  __bf16* Gbf = (__bf16*)(ws + off);
  off += (size_t)C_ROWS * DIM * sizeof(__bf16);
  off = (off + 255) & ~(size_t)255;
  float* cand_val = (float*)(ws + off);
  off += (size_t)B_ROWS * NSEG * TOPK * sizeof(float);
  int* cand_idx = (int*)(ws + off);

  nrm_bf16_kernel<<<B_ROWS, 256, 0, stream>>>(emb, Abf);
  nrm_bf16_kernel<<<C_ROWS, 256, 0, stream>>>(idx, Gbf);

  const size_t smem = (size_t)MBLOCK * ASTRIDE * sizeof(__bf16)        // A tile   (~66.5 KB)
                    + (size_t)NSTRIP * MTILES * 256 * sizeof(float)    // staging  (64 KB)
                    + (size_t)MBLOCK * TOPK * (sizeof(float) + sizeof(int))  // top-k (51.2 KB)
                    + (size_t)MBLOCK * (sizeof(float) + sizeof(int));  // ~184 KB (<320 KB WGP LDS)
  gemm_topk_kernel<<<dim3(B_ROWS / MBLOCK, NSEG), 256, smem, stream>>>(
      Abf, Gbf, cand_val, cand_idx);

  float* out_val = (float*)d_out;
  int*   out_idx = (int*)((float*)d_out + (size_t)B_ROWS * TOPK);
  merge_topk_kernel<<<B_ROWS, 256, 0, stream>>>(cand_val, cand_idx, out_val, out_idx);
}
